// RefAttention_83932250899032
// MI455X (gfx1250) — matrix operand order repaired and checked by reading, not compiled, not run on hardware
//
#include <hip/hip_runtime.h>

// ---------------------------------------------------------------------------
// RefAttention for MI455X (gfx1250, wave32, WMMA f32_16x16x32_f16)
//   B=8, N=1024, C=640, H=8, D=80 (stored padded to 96), R=3 -> 3072 keys
// Stage 1: QKV projection  (f32 in -> f16 q/k/v [B,H,N,96], d>=80 zeroed)
// Stage 2: flash attention per (b,h): async-to-LDS K/V staging, f16 WMMA,
//          f32 accum, online softmax
// Stage 3: output projection (f16 att x f32 Wo -> f32 out + bias)
// ---------------------------------------------------------------------------

typedef __attribute__((ext_vector_type(16))) _Float16 v16h;
typedef __attribute__((ext_vector_type(8)))  float    v8f;

#define BATCH 8
#define NTOK  1024
#define CDIM  640
#define HEADS 8
#define HD    80
#define DP    96              // padded head dim (3 x 32 for WMMA K)
#define RWIN  3
#define NKEYS (RWIN * NTOK)   // 3072

#if defined(__HIP_DEVICE_COMPILE__)
typedef int v4i __attribute__((vector_size(16)));
typedef __attribute__((address_space(1))) v4i* gas_v4i;   // global (AS1)
typedef __attribute__((address_space(3))) v4i* lds_v4i;   // LDS (AS3)
#endif

__device__ __forceinline__ v8f v8f_zero() {
  v8f z = {0.f, 0.f, 0.f, 0.f, 0.f, 0.f, 0.f, 0.f};
  return z;
}

// A-fragment pair->K mapping for 16x32 f16 (contiguous k = 16*lh + 2j): lane half lh in {0,1}
__device__ __forceinline__ int a_kpair(int j, int lh) {
  return lh * 16 + 2 * j;
}

// ---------------------------------------------------------------------------
// Stage 1: q/k/v = x @ W^T  (W is [inner=640, C=640], f32)
// grid = (512 row-tiles, 3 {q,k,v}), block = 256 (8 waves)
// wave w owns column tiles ct = w + 8*t, t=0..4  (40 tiles of 16 over inner)
// Output layout [B][H][N][96]; d in [80,96) zero-filled here.
// ---------------------------------------------------------------------------
__global__ __launch_bounds__(256) void qkv_kernel(
    const float* __restrict__ x,
    const float* __restrict__ Wq,
    const float* __restrict__ Wk,
    const float* __restrict__ Wv,
    _Float16* __restrict__ q,
    _Float16* __restrict__ k,
    _Float16* __restrict__ v) {
  const int wave = threadIdx.x >> 5;
  const int lane = threadIdx.x & 31;
  const int lh = lane >> 4, ll = lane & 15;
  const int mbase = blockIdx.x * 16;
  const int which = blockIdx.y;

  const float* W = (which == 0) ? Wq : (which == 1) ? Wk : Wv;
  _Float16* outp = (which == 0) ? q : (which == 1) ? k : v;

  v8f acc[5];
#pragma unroll
  for (int t = 0; t < 5; ++t) acc[t] = v8f_zero();

  const float* xrow0 = x + (size_t)(mbase + ll) * CDIM;

  for (int kc = 0; kc < CDIM; kc += 32) {
    // A fragment (x, f32 -> f16), 16 rows x 32 k
    v16h a;
#pragma unroll
    for (int j = 0; j < 8; ++j) {
      const int kk = a_kpair(j, lh);
      float2 f = *(const float2*)(xrow0 + kc + kk);
      a[2 * j]     = (_Float16)f.x;
      a[2 * j + 1] = (_Float16)f.y;
    }
#pragma unroll
    for (int t = 0; t < 5; ++t) {
      const int colbase = (wave + 8 * t) * 16;
      // B fragment: B[k0][n] = W[colbase+n][kc+k0]  (32 k x 16 n)
      v16h bf;
      const float* wrow = W + (size_t)(colbase + ll) * CDIM + kc;
#pragma unroll
      for (int j = 0; j < 8; ++j) {
        const int k0 = lh * 16 + 2 * j;
        float2 f = *(const float2*)(wrow + k0);
        bf[2 * j]     = (_Float16)f.x;
        bf[2 * j + 1] = (_Float16)f.y;
      }
      acc[t] = __builtin_amdgcn_wmma_f32_16x16x32_f16(
          false, a, false, bf, (short)0, acc[t], false, false);
    }
  }

  // Store: [B][H][N][96]; a 16-col tile lies within one head (80 = 5*16)
#pragma unroll
  for (int t = 0; t < 5; ++t) {
    const int ct = wave + 8 * t;
    const int h = ct / 5, dbase = (ct % 5) * 16;
#pragma unroll
    for (int r = 0; r < 8; ++r) {
      const int mrow = mbase + r + 8 * lh;
      const int bb = mrow >> 10, nn = mrow & (NTOK - 1);
      const size_t off =
          ((((size_t)bb * HEADS + h) * NTOK) + nn) * DP + dbase + ll;
      outp[off] = (_Float16)acc[t][r];
    }
  }

  // Zero-fill pad region d in [80,96) for this block's 16 rows (all heads)
  for (int i = threadIdx.x; i < 16 * HEADS * 16; i += 256) {  // 2048
    const int row = i >> 7;         // /128
    const int rem = i & 127;
    const int h = rem >> 4;
    const int d = HD + (rem & 15);
    const int mrow = mbase + row;
    const int bb = mrow >> 10, nn = mrow & (NTOK - 1);
    outp[((((size_t)bb * HEADS + h) * NTOK) + nn) * DP + d] = (_Float16)0.f;
  }
}

// ---------------------------------------------------------------------------
// Stage 2: flash attention. grid = B*H*8 = 512 blocks, block = 256 (8 waves).
// Each wave: 16 query rows; block iterates 96 tiles of 32 keys.
// K/V tiles are contiguous 6144-byte blocks -> async b128 copies to LDS.
// ---------------------------------------------------------------------------
__global__ __launch_bounds__(256) void attn_kernel(
    const _Float16* __restrict__ q,
    const _Float16* __restrict__ k,
    const _Float16* __restrict__ v,
    const int* __restrict__ ctx_idx,   // [B][3]
    _Float16* __restrict__ att) {      // [B][N][640] (i = h*80 + d)
  __shared__ __align__(16) _Float16 kls[32 * DP];    // [key][d]
  __shared__ __align__(16) _Float16 vls[32 * DP];    // [key][d]
  __shared__ _Float16 pls[8][16 * 32];               // per-wave P transpose

  const int wave = threadIdx.x >> 5;
  const int lane = threadIdx.x & 31;
  const int lh = lane >> 4, ll = lane & 15;

  const int bh = blockIdx.x >> 3;
  const int b = bh >> 3, h = bh & (HEADS - 1);
  const int qbase = (blockIdx.x & 7) * 128 + wave * 16;

  const float scale = 0.11180339887498948f; // 80^-0.5

  // Q fragments: 3 chunks of 32 d, pre-scaled (pad d>=80 already zero)
  v16h qa[3];
  {
    const _Float16* qrow =
        q + ((((size_t)b * HEADS + h) * NTOK) + (qbase + ll)) * DP;
#pragma unroll
    for (int c0 = 0; c0 < 3; ++c0) {
#pragma unroll
      for (int j = 0; j < 8; ++j) {
        const int kk = c0 * 32 + a_kpair(j, lh);
        qa[c0][2 * j]     = (_Float16)((float)qrow[kk] * scale);
        qa[c0][2 * j + 1] = (_Float16)((float)qrow[kk + 1] * scale);
      }
    }
  }

  v8f acc[6];
#pragma unroll
  for (int t = 0; t < 6; ++t) acc[t] = v8f_zero();
  float mrun[8], lrun[8];
#pragma unroll
  for (int r = 0; r < 8; ++r) { mrun[r] = -1e30f; lrun[r] = 0.f; }

  const int f0 = ctx_idx[b * RWIN + 0];
  const int f1 = ctx_idx[b * RWIN + 1];
  const int f2 = ctx_idx[b * RWIN + 2];

  for (int kt = 0; kt < NKEYS / 32; ++kt) {
    const int key0 = kt * 32;
    const int fr = key0 >> 10;
    const int frame = (fr == 0) ? f0 : (fr == 1) ? f1 : f2;
    const int tok0 = key0 & (NTOK - 1);
    const _Float16* kp = k + (((size_t)frame * HEADS + h) * NTOK + tok0) * DP;
    const _Float16* vp = v + (((size_t)frame * HEADS + h) * NTOK + tok0) * DP;

    __syncthreads();   // previous tile fully consumed before overwrite
#if __has_builtin(__builtin_amdgcn_global_load_async_to_lds_b128)
    {
      // 32 keys x 192B contiguous = 6144B = 384 b128 transfers per matrix
      const char* kg = (const char*)kp;
      const char* vg = (const char*)vp;
      char* kl = (char*)kls;
      char* vl = (char*)vls;
      for (int i = threadIdx.x; i < 384; i += 256) {
        __builtin_amdgcn_global_load_async_to_lds_b128(
            (gas_v4i)(kg + (size_t)i * 16), (lds_v4i)(kl + (size_t)i * 16),
            0, 0);
        __builtin_amdgcn_global_load_async_to_lds_b128(
            (gas_v4i)(vg + (size_t)i * 16), (lds_v4i)(vl + (size_t)i * 16),
            0, 0);
      }
      asm volatile("s_wait_asynccnt 0x0" ::: "memory");
    }
#else
    for (int idx = threadIdx.x; idx < 32 * DP; idx += 256) {
      kls[idx] = kp[idx];
      vls[idx] = vp[idx];
    }
#endif
    __syncthreads();

    if (kt + 1 < NKEYS / 32) {   // prefetch next K tile -> global_prefetch_b8
      const int nk0 = (kt + 1) * 32;
      const int nfr = nk0 >> 10;
      const int nframe = (nfr == 0) ? f0 : (nfr == 1) ? f1 : f2;
      const _Float16* np =
          k + (((size_t)nframe * HEADS + h) * NTOK + (nk0 & (NTOK - 1))) * DP;
      __builtin_prefetch(np, 0, 1);
    }

    // Scores: two 16x16 tiles (keys 0-15, 16-31), K-reduction over 96 (3 WMMA)
    v8f s0 = v8f_zero(), s1 = v8f_zero();
#pragma unroll
    for (int c0 = 0; c0 < 3; ++c0) {
      v16h b0, b1;
#pragma unroll
      for (int j = 0; j < 8; ++j) {
        const int kk = c0 * 32 + lh * 16 + 2 * j;   // d index (WMMA k)
        b0[2 * j]     = kls[ll * DP + kk];
        b0[2 * j + 1] = kls[ll * DP + kk + 1];
        b1[2 * j]     = kls[(16 + ll) * DP + kk];
        b1[2 * j + 1] = kls[(16 + ll) * DP + kk + 1];
      }
      s0 = __builtin_amdgcn_wmma_f32_16x16x32_f16(
          false, qa[c0], false, b0, (short)0, s0, false, false);
      s1 = __builtin_amdgcn_wmma_f32_16x16x32_f16(
          false, qa[c0], false, b1, (short)0, s1, false, false);
    }

    // Online softmax (row stats replicated over each 16-lane group)
    float p0[8], p1[8], sf[8];
#pragma unroll
    for (int r = 0; r < 8; ++r) {
      float mx = fmaxf(s0[r], s1[r]);
#pragma unroll
      for (int off = 8; off >= 1; off >>= 1)
        mx = fmaxf(mx, __shfl_xor(mx, off, 32));
      const float mnew = fmaxf(mrun[r], mx);
      sf[r] = __expf(mrun[r] - mnew);
      p0[r] = __expf(s0[r] - mnew);
      p1[r] = __expf(s1[r] - mnew);
      float rs = p0[r] + p1[r];
#pragma unroll
      for (int off = 8; off >= 1; off >>= 1)
        rs += __shfl_xor(rs, off, 32);
      lrun[r] = lrun[r] * sf[r] + rs;
      mrun[r] = mnew;
    }
#pragma unroll
    for (int t = 0; t < 6; ++t)
#pragma unroll
      for (int r = 0; r < 8; ++r) acc[t][r] *= sf[r];

    // Transpose P (C-layout -> A-fragment) through wave-private LDS
    _Float16* pw = &pls[wave][0];
#pragma unroll
    for (int r = 0; r < 8; ++r) {
      pw[(r + 8 * lh) * 32 + ll]      = (_Float16)p0[r];
      pw[(r + 8 * lh) * 32 + 16 + ll] = (_Float16)p1[r];
    }
    asm volatile("s_wait_dscnt 0x0" ::: "memory"); // same-wave LDS RAW
    v16h pa;
#pragma unroll
    for (int j = 0; j < 8; ++j) {
      const int kk = a_kpair(j, lh);
      pa[2 * j]     = pw[ll * 32 + kk];
      pa[2 * j + 1] = pw[ll * 32 + kk + 1];
    }

    // P (16x32 keys) x V (32 keys x 16 d) per d-column tile
#pragma unroll
    for (int t = 0; t < 6; ++t) {
      v16h bv;
#pragma unroll
      for (int j = 0; j < 8; ++j) {
        const int kk = lh * 16 + 2 * j;             // key index (WMMA k)
        bv[2 * j]     = vls[kk * DP + t * 16 + ll];
        bv[2 * j + 1] = vls[(kk + 1) * DP + t * 16 + ll];
      }
      acc[t] = __builtin_amdgcn_wmma_f32_16x16x32_f16(
          false, pa, false, bv, (short)0, acc[t], false, false);
    }
  }

  // Epilogue: normalize and store att[B][N][h*80+d] (f16)
#pragma unroll
  for (int r = 0; r < 8; ++r) {
    const float inv = 1.0f / lrun[r];
    const int nq = qbase + r + 8 * lh;
#pragma unroll
    for (int t = 0; t < 6; ++t) {
      const int d = t * 16 + ll;
      if (d < HD) {
        att[((size_t)b * NTOK + nq) * CDIM + h * HD + d] =
            (_Float16)(acc[t][r] * inv);
      }
    }
  }
}

// ---------------------------------------------------------------------------
// Stage 3: out = att @ Wo^T + bo  (att f16 [8192,640], Wo f32 [640,640])
// grid = 512 row tiles, block = 256 (8 waves x 5 column tiles)
// ---------------------------------------------------------------------------
__global__ __launch_bounds__(256) void oproj_kernel(
    const _Float16* __restrict__ att,
    const float* __restrict__ Wo,
    const float* __restrict__ bo,
    float* __restrict__ out) {
  const int wave = threadIdx.x >> 5;
  const int lane = threadIdx.x & 31;
  const int lh = lane >> 4, ll = lane & 15;
  const int mbase = blockIdx.x * 16;

  v8f acc[5];
#pragma unroll
  for (int t = 0; t < 5; ++t) acc[t] = v8f_zero();

  const _Float16* arow = att + (size_t)(mbase + ll) * CDIM;

  for (int kc = 0; kc < CDIM; kc += 32) {
    v16h a;
#pragma unroll
    for (int j = 0; j < 8; ++j) {
      const int kk = kc + a_kpair(j, lh);
      a[2 * j]     = arow[kk];
      a[2 * j + 1] = arow[kk + 1];
    }
#pragma unroll
    for (int t = 0; t < 5; ++t) {
      const int colbase = (wave + 8 * t) * 16;
      v16h bf;
      const float* wrow = Wo + (size_t)(colbase + ll) * CDIM + kc;
#pragma unroll
      for (int j = 0; j < 8; ++j) {
        const int k0 = lh * 16 + 2 * j;
        float2 f = *(const float2*)(wrow + k0);
        bf[2 * j]     = (_Float16)f.x;
        bf[2 * j + 1] = (_Float16)f.y;
      }
      acc[t] = __builtin_amdgcn_wmma_f32_16x16x32_f16(
          false, a, false, bf, (short)0, acc[t], false, false);
    }
  }

#pragma unroll
  for (int t = 0; t < 5; ++t) {
    const int colbase = (wave + 8 * t) * 16;
    const float bias = bo[colbase + ll];
#pragma unroll
    for (int r = 0; r < 8; ++r) {
      const int mrow = mbase + r + 8 * lh;
      out[(size_t)mrow * CDIM + colbase + ll] = acc[t][r] + bias;
    }
  }
}

// ---------------------------------------------------------------------------
extern "C" void kernel_launch(void* const* d_in, const int* in_sizes, int n_in,
                              void* d_out, int out_size, void* d_ws,
                              size_t ws_size, hipStream_t stream) {
  const float* x  = (const float*)d_in[0];
  const float* Wq = (const float*)d_in[1];
  const float* Wk = (const float*)d_in[2];
  const float* Wv = (const float*)d_in[3];
  const float* Wo = (const float*)d_in[4];
  const float* bo = (const float*)d_in[5];
  const int*  ctx = (const int*)d_in[6];
  float* out = (float*)d_out;

  // Workspace layout (f16): q,k,v = [B,H,N,96] (12 MiB each), att = [B,N,640]
  char* ws = (char*)d_ws;
  const size_t headBytes = (size_t)BATCH * HEADS * NTOK * DP * 2; // 12.6 MB
  _Float16* qws = (_Float16*)(ws);
  _Float16* kws = (_Float16*)(ws + headBytes);
  _Float16* vws = (_Float16*)(ws + 2 * headBytes);
  _Float16* att = (_Float16*)(ws + 3 * headBytes);

  qkv_kernel<<<dim3(512, 3), 256, 0, stream>>>(x, Wq, Wk, Wv, qws, kws, vws);
  attn_kernel<<<512, 256, 0, stream>>>(qws, kws, vws, ctx, att);
  oproj_kernel<<<512, 256, 0, stream>>>(att, Wo, bo, out);
}
